// InfoNCELoss_2456721293347
// MI455X (gfx1250) — matrix operand order repaired and checked by reading, not compiled, not run on hardware
//
#include <hip/hip_runtime.h>
#include <hip/hip_bf16.h>

typedef __attribute__((ext_vector_type(16))) _Float16 v16h;
typedef __attribute__((ext_vector_type(8)))  _Float16 v8h;
typedef __attribute__((ext_vector_type(8)))  float    v8f;

#define B_ROWS 8192
#define DIMS   256
#define NTILE  (B_ROWS / 16)          // 512 column tiles
#define INV_T  14.285714285714286f    // 1 / 0.07  (also the fixed LSE shift C)

// ---------------- kernel 0: zero the accumulators ----------------
__global__ void init_kernel(float* acc) {
    acc[0] = 0.0f;   // sum of per-row losses
    acc[1] = 0.0f;   // count of rows with positives
}

// ---------------- kernel 1: L2-normalize rows, f32 -> f16 ----------------
// 8 waves per block, one wave per row; lane handles 8 contiguous floats.
__global__ void __launch_bounds__(256) normalize_kernel(const float* __restrict__ x,
                                                        _Float16* __restrict__ eh) {
    const int lane = threadIdx.x & 31;
    const int wave = threadIdx.x >> 5;
    const int row  = blockIdx.x * 8 + wave;

    const float* p = x + (size_t)row * DIMS + lane * 8;
    float4 a = *(const float4*)p;
    float4 b = *(const float4*)(p + 4);
    float ss = a.x * a.x + a.y * a.y + a.z * a.z + a.w * a.w
             + b.x * b.x + b.y * b.y + b.z * b.z + b.w * b.w;
#pragma unroll
    for (int off = 16; off >= 1; off >>= 1)
        ss += __shfl_xor(ss, off, 32);

    const float inv = 1.0f / fmaxf(sqrtf(ss), 1e-12f);
    v8h o;
    o[0] = (_Float16)(a.x * inv); o[1] = (_Float16)(a.y * inv);
    o[2] = (_Float16)(a.z * inv); o[3] = (_Float16)(a.w * inv);
    o[4] = (_Float16)(b.x * inv); o[5] = (_Float16)(b.y * inv);
    o[6] = (_Float16)(b.z * inv); o[7] = (_Float16)(b.w * inv);
    *(v8h*)(eh + (size_t)row * DIMS + lane * 8) = o;
}

// ---------------- kernel 2: fused WMMA GEMM + masked fixed-shift softmax sums ----
// One block (4 waves) per 16-row tile; wave w sweeps column tiles j = w, w+4, ...
// Each tile: 8x V_WMMA_F32_16X16X32_F16 (K=256), then a branchless epilogue
// accumulating sum(exp(sim/T - C)) per row for the "all but diag" mask and the
// "same label, not diag" mask.  C = 1/T >= max possible sim/T, so no running
// max is needed and C cancels in the final loss: loss = log(sAll) - log(sPos).
__global__ void __launch_bounds__(128)
supcon_kernel(const _Float16* __restrict__ eh,
              const int* __restrict__ labels,
              float* __restrict__ acc) {
    __shared__ float lds[4][16][2];

    const int lane = threadIdx.x & 31;
    const int wave = threadIdx.x >> 5;
    const int l15  = lane & 15;
    const int hi   = lane >> 4;          // 0: low half, 1: high half of wave
    const int rowBase = blockIdx.x * 16;

    // ---- A fragments: 16 rows x 256 K, ISA 16-bit A layout ----
    const int mrow = rowBase + l15;
    const int koff = hi ? 8 : 0;
    const _Float16* arp = eh + (size_t)mrow * DIMS;
    v16h a[8];
#pragma unroll
    for (int kk = 0; kk < 8; ++kk) {
        v8h lo  = *(const v8h*)(arp + kk * 32 + koff);
        v8h hi8 = *(const v8h*)(arp + kk * 32 + 16 + koff);
        a[kk] = __builtin_shufflevector(lo, hi8, 0, 1, 2, 3, 4, 5, 6, 7,
                                                 8, 9, 10, 11, 12, 13, 14, 15);
    }

    // Labels and global row ids of the 8 C-slots this lane owns (row = vg + 8*hi)
    int labr[8];
#pragma unroll
    for (int vg = 0; vg < 8; ++vg)
        labr[vg] = labels[rowBase + vg + hi * 8];

    float sAll[8], sPos[8];
#pragma unroll
    for (int vg = 0; vg < 8; ++vg) { sAll[vg] = 0.0f; sPos[vg] = 0.0f; }

    // ---- sweep column tiles (uniform per wave -> EXEC all-1s at WMMA) ----
    for (int j = wave; j < NTILE; j += 4) {
        const int col  = j * 16 + l15;
        const int labc = labels[col];

        // B fragment: column N (= row `col` of eh, since B = e^T), table-order K (same as A).
        const _Float16* brp = eh + (size_t)col * DIMS + (hi ? 8 : 0);

        v8f c = {0.f, 0.f, 0.f, 0.f, 0.f, 0.f, 0.f, 0.f};
#pragma unroll
        for (int kk = 0; kk < 8; ++kk) {
            v8h lo  = *(const v8h*)(brp + kk * 32);
            v8h hi8 = *(const v8h*)(brp + kk * 32 + 16);
            v16h b = __builtin_shufflevector(lo, hi8, 0, 1, 2, 3, 4, 5, 6, 7,
                                                      8, 9, 10, 11, 12, 13, 14, 15);
            c = __builtin_amdgcn_wmma_f32_16x16x32_f16(
                    false, a[kk], false, b, (short)0, c, false, false);
        }

        // ---- branchless epilogue: e = exp((c - 1) / T), masked accumulate ----
#pragma unroll
        for (int vg = 0; vg < 8; ++vg) {
            const int r = rowBase + vg + hi * 8;
            const float e    = __expf(__builtin_fmaf(c[vg], INV_T, -INV_T));
            const float eAll = (col != r) ? e : 0.0f;        // exclude diagonal
            const float ePos = (labc == labr[vg]) ? eAll : 0.0f;
            sAll[vg] += eAll;
            sPos[vg] += ePos;
        }
    }

    // ---- cross-lane sum within each 16-lane half ----
#pragma unroll
    for (int vg = 0; vg < 8; ++vg) {
#pragma unroll
        for (int off = 1; off < 16; off <<= 1) {
            sAll[vg] += __shfl_xor(sAll[vg], off, 16);
            sPos[vg] += __shfl_xor(sPos[vg], off, 16);
        }
    }

    // ---- merge the 4 waves' partials via LDS ----
    if (l15 == 0) {
#pragma unroll
        for (int vg = 0; vg < 8; ++vg) {
            const int row = vg + hi * 8;
            lds[wave][row][0] = sAll[vg];
            lds[wave][row][1] = sPos[vg];
        }
    }
    __syncthreads();

    if (threadIdx.x < 16) {
        const int row = threadIdx.x;
        float sa = lds[0][row][0] + lds[1][row][0] + lds[2][row][0] + lds[3][row][0];
        float sp = lds[0][row][1] + lds[1][row][1] + lds[2][row][1] + lds[3][row][1];
        float loss = 0.0f, cnt = 0.0f;
        if (sp > 0.0f) {                 // row has >= 1 positive
            loss = __logf(sa) - __logf(sp);   // fixed shift C cancels
            cnt  = 1.0f;
        }
        atomicAdd(&acc[0], loss);
        atomicAdd(&acc[1], cnt);
    }
}

// ---------------- kernel 3: finalize scalar ----------------
__global__ void final_kernel(const float* __restrict__ acc, float* __restrict__ out) {
    const float c = acc[1];
    out[0] = (c > 0.0f) ? (acc[0] / c) : 0.0f;
}

extern "C" void kernel_launch(void* const* d_in, const int* in_sizes, int n_in,
                              void* d_out, int out_size, void* d_ws, size_t ws_size,
                              hipStream_t stream) {
    const float* x      = (const float*)d_in[0];   // embeddings [8192, 256] f32
    const int*   labels = (const int*)d_in[1];     // labels [8192] int

    float*    acc = (float*)d_ws;                          // 2 accumulators
    _Float16* eh  = (_Float16*)((char*)d_ws + 64);         // normalized f16 [8192,256]

    init_kernel<<<1, 1, 0, stream>>>(acc);
    normalize_kernel<<<B_ROWS / 8, 256, 0, stream>>>(x, eh);
    supcon_kernel<<<B_ROWS / 16, 128, 0, stream>>>(eh, labels, acc);
    final_kernel<<<1, 1, 0, stream>>>(acc, (float*)d_out);
}
